// OurGMN_67912022884661
// MI455X (gfx1250) — matrix operand order repaired and checked by reading, not compiled, not run on hardware
//
#include <hip/hip_runtime.h>
#include <hip/hip_fp16.h>

typedef __attribute__((ext_vector_type(16))) _Float16 v16h;
typedef __attribute__((ext_vector_type(8)))  _Float16 h8v;
typedef __attribute__((ext_vector_type(8)))  float    v8f;

// ---- WMMA fragment index maps (CDNA5 ISA 7.12.2, wave32) ----
// A 16x32 f16: lanes 0-15 row M=lane, K = {0..7,16..23}; lanes 16-31 same row, K = {8..15,24..31}
// B 32x16 f16: lanes 0-15 col N=lane&15 hold K = {0..7,16..23}; lanes 16-31 hold K = {8..15,24..31}
__device__ __forceinline__ int kmapB(int i, int lane) { return (i & 7) + ((lane >> 4) << 3) + ((i >> 3) << 4); }

// Unconditional tile store: rows are guaranteed multiples of 16 in this problem,
// so all 32 stores are immediate-offset b32 off one per-lane base address.
template<bool RELU, bool MIRROR>
__device__ __forceinline__
void store_tile(v8f (&acc)[4], float* __restrict__ C, _Float16* __restrict__ Ch,
                int row0, int lane)
{
    int mbase = row0 + ((lane >> 4) << 3);
    int coln  = lane & 15;
    float*    p  = C + (size_t)mbase * 64 + coln;
    _Float16* ph = Ch + (size_t)mbase * 64 + coln;
#pragma unroll
    for (int t = 0; t < 4; ++t)
#pragma unroll
        for (int j = 0; j < 8; ++j) {
            float v = acc[t][j];
            if (RELU) v = fmaxf(v, 0.0f);
            p[j * 64 + t * 16] = v;
            if (MIRROR) ph[j * 64 + t * 16] = (_Float16)v;
        }
}

// C[rows x 64] = act(A[rows x K] @ W[K x 64] + bias); optional f16 mirror Ch.
// A f16 row-major; W f32 row-major, pre-swizzled into LDS in B-fragment order.
// rows must be a multiple of 16. K=64: all B fragments persist in VGPRs across
// a grid-stride row-tile loop (steady state: 4 global_load_b128 + 8 wmma).
template<int K, bool RELU, bool MIRROR>
__global__ __launch_bounds__(256)
void gemm_n64(const _Float16* __restrict__ A, const float* __restrict__ W,
              const float* __restrict__ bias, float* __restrict__ C,
              _Float16* __restrict__ Ch, int rows)
{
    constexpr int KT = K / 32;
    __shared__ v16h  Wf[KT * 4 * 32];
    __shared__ float bl[64];
    int tid = threadIdx.x;
    _Float16* Wh = (_Float16*)Wf;
    for (int idx = tid; idx < K * 64; idx += 256) {
        int i  = idx & 15;
        int ln = (idx >> 4) & 31;
        int t  = (idx >> 9) & 3;
        int kt = idx >> 11;
        Wh[idx] = (_Float16)W[((kt << 5) + kmapB(i, ln)) * 64 + t * 16 + (ln & 15)];
    }
    if (tid < 64) bl[tid] = bias ? bias[tid] : 0.0f;
    __syncthreads();

    int wave = tid >> 5, lane = tid & 31;
    int base = (lane >> 4) << 3;                 // A fragment K base: 0 or 8
    float bv[4];
#pragma unroll
    for (int t = 0; t < 4; ++t) bv[t] = bl[t * 16 + (lane & 15)];

    int ntile = rows >> 4;

    if constexpr (K == 64) {
        // hoist all 8 B fragments into registers once per wave
        v16h b[2][4];
#pragma unroll
        for (int kt = 0; kt < 2; ++kt)
#pragma unroll
            for (int t = 0; t < 4; ++t) b[kt][t] = Wf[kt * 128 + t * 32 + lane];

        for (int tile = blockIdx.x * 8 + wave; tile < ntile; tile += gridDim.x * 8) {
            int row0 = tile << 4;
            const _Float16* rowp = A + (size_t)(row0 + (lane & 15)) * K + base;
            v16h a[2];
#pragma unroll
            for (int kt = 0; kt < 2; ++kt) {
                const h8v* pa = (const h8v*)(rowp + kt * 32);
                a[kt] = __builtin_shufflevector(pa[0], pa[2], 0, 1, 2, 3, 4, 5, 6, 7,
                                                             8, 9, 10, 11, 12, 13, 14, 15);
            }
            v8f acc[4];
#pragma unroll
            for (int t = 0; t < 4; ++t)
#pragma unroll
                for (int j = 0; j < 8; ++j) acc[t][j] = bv[t];
#pragma unroll
            for (int kt = 0; kt < 2; ++kt)
#pragma unroll
                for (int t = 0; t < 4; ++t)
                    acc[t] = __builtin_amdgcn_wmma_f32_16x16x32_f16(
                        false, a[kt], false, b[kt][t], (short)0, acc[t], false, false);
            store_tile<RELU, MIRROR>(acc, C, Ch, row0, lane);
        }
    } else {
        for (int tile = blockIdx.x * 8 + wave; tile < ntile; tile += gridDim.x * 8) {
            int row0 = tile << 4;
            const _Float16* rowp = A + (size_t)(row0 + (lane & 15)) * K + base;
            v8f acc[4];
#pragma unroll
            for (int t = 0; t < 4; ++t)
#pragma unroll
                for (int j = 0; j < 8; ++j) acc[t][j] = bv[t];
#pragma unroll
            for (int kt = 0; kt < KT; ++kt) {
                const h8v* pa = (const h8v*)(rowp + kt * 32);
                v16h a = __builtin_shufflevector(pa[0], pa[2], 0, 1, 2, 3, 4, 5, 6, 7,
                                                              8, 9, 10, 11, 12, 13, 14, 15);
                const v16h* wrow = Wf + kt * 128 + lane;
#pragma unroll
                for (int t = 0; t < 4; ++t)
                    acc[t] = __builtin_amdgcn_wmma_f32_16x16x32_f16(
                        false, a, false, wrow[t * 32], (short)0, acc[t], false, false);
            }
            store_tile<RELU, MIRROR>(acc, C, Ch, row0, lane);
        }
    }
}

// Cross attention: one wave per target node t. logit(q,t)=relu(HqA[q]+HtA[t])·w2a
// (logit bias cancels in the softmax over q); Hbar[t]=sum_q a_q*relu(HqV[q]+HtV[t]).
// Writes Hbar in f16 (it feeds the next WMMA GEMM directly).
__global__ __launch_bounds__(256)
void cross_attention_kernel(const float* __restrict__ HqA, const float* __restrict__ HqV,
                            const float* __restrict__ HtA, const float* __restrict__ HtV,
                            const float* __restrict__ w2a,
                            const unsigned char* __restrict__ mask, // [16][Nt] bool
                            _Float16* __restrict__ Hbar, int Nt)
{
    __shared__ float sA[1024], sV[1024], sw[64];
    int tid = threadIdx.x;
    for (int i = tid; i < 1024; i += blockDim.x) { sA[i] = HqA[i]; sV[i] = HqV[i]; }
    if (tid < 64) sw[tid] = w2a[tid];
    __syncthreads();

    int wave = tid >> 5, lane = tid & 31;
    int t = blockIdx.x * 8 + wave;
    if (t >= Nt) return;

    int j0 = lane, j1 = lane + 32;
    float ha0 = HtA[(size_t)t * 64 + j0], ha1 = HtA[(size_t)t * 64 + j1];
    float hv0 = HtV[(size_t)t * 64 + j0], hv1 = HtV[(size_t)t * 64 + j1];
    float w0 = sw[j0], w1 = sw[j1];

    float logit[16];
#pragma unroll
    for (int q = 0; q < 16; ++q) {
        float p = fmaxf(sA[q * 64 + j0] + ha0, 0.0f) * w0
                + fmaxf(sA[q * 64 + j1] + ha1, 0.0f) * w1;
        for (int off = 16; off > 0; off >>= 1) p += __shfl_xor(p, off, 32);
        logit[q] = p;
    }
    float mx = -3.4e38f;
#pragma unroll
    for (int q = 0; q < 16; ++q) {
        bool al = mask[(size_t)q * Nt + t] != 0;
        logit[q] = al ? logit[q] : -3.4e38f;
        mx = fmaxf(mx, logit[q]);
    }
    float aq[16], denom = 0.0f;
#pragma unroll
    for (int q = 0; q < 16; ++q) {
        aq[q] = (logit[q] > -1.0e37f) ? __expf(logit[q] - mx) : 0.0f;
        denom += aq[q];
    }
    float inv = 1.0f / denom;
    float o0 = 0.0f, o1 = 0.0f;
#pragma unroll
    for (int q = 0; q < 16; ++q) {
        float a = aq[q] * inv;
        o0 += a * fmaxf(sV[q * 64 + j0] + hv0, 0.0f);
        o1 += a * fmaxf(sV[q * 64 + j1] + hv1, 0.0f);
    }
    Hbar[(size_t)t * 64 + j0] = (_Float16)o0;
    Hbar[(size_t)t * 64 + j1] = (_Float16)o1;
}

__global__ void colsum64(const float* __restrict__ X, float* __restrict__ Q, int rows) {
    __shared__ float s[256];
    int tid = threadIdx.x, col = tid & 63, part = tid >> 6;
    float acc = 0.0f;
    for (int r = part; r < rows; r += 4) acc += X[(size_t)r * 64 + col];
    s[tid] = acc;
    __syncthreads();
    if (tid < 64) Q[tid] = s[tid] + s[tid + 64] + s[tid + 128] + s[tid + 192];
}

__global__ void build_xtm(const float* __restrict__ Xt, const float* __restrict__ Xqt,
                          const float* __restrict__ Q, _Float16* __restrict__ XtM, int Nt) {
    size_t idx = (size_t)blockIdx.x * blockDim.x + threadIdx.x;
    size_t total = (size_t)Nt * 64, st = (size_t)gridDim.x * blockDim.x;
    for (; idx < total; idx += st) {
        size_t n = idx >> 6; int j = (int)(idx & 63);
        XtM[n * 192 + j]       = (_Float16)Xt[idx];
        XtM[n * 192 + 64 + j]  = (_Float16)Xqt[idx];
        XtM[n * 192 + 128 + j] = (_Float16)Q[j];
    }
}

__global__ void f32_to_f16(const float* __restrict__ src, _Float16* __restrict__ dst, size_t n) {
    size_t i = (size_t)blockIdx.x * blockDim.x + threadIdx.x;
    size_t st = (size_t)gridDim.x * blockDim.x;
    for (; i < n; i += st) dst[i] = (_Float16)src[i];
}

// One wave per edge: att=sigmoid(relu(A1[s]+A2[d])·w2 + b2); out[d] += att*V[s]
__global__ __launch_bounds__(256)
void edge_msg(const int* __restrict__ src, const int* __restrict__ dst,
              const float* __restrict__ A1, const float* __restrict__ A2,
              const float* __restrict__ V, const float* __restrict__ w2,
              const float* __restrict__ b2, float* __restrict__ out, int E)
{
    int wave = threadIdx.x >> 5, lane = threadIdx.x & 31;
    int e = blockIdx.x * 8 + wave;
    if (e >= E) return;
    int s = src[e], d = dst[e];
    int j0 = lane, j1 = lane + 32;
    float h0 = fmaxf(A1[(size_t)s * 64 + j0] + A2[(size_t)d * 64 + j0], 0.0f);
    float h1 = fmaxf(A1[(size_t)s * 64 + j1] + A2[(size_t)d * 64 + j1], 0.0f);
    float p = h0 * w2[j0] + h1 * w2[j1];
    for (int off = 16; off > 0; off >>= 1) p += __shfl_xor(p, off, 32);
    float att = 1.0f / (1.0f + __expf(-(p + b2[0])));
    atomicAdd(&out[(size_t)d * 64 + j0], att * V[(size_t)s * 64 + j0]);
    atomicAdd(&out[(size_t)d * 64 + j1], att * V[(size_t)s * 64 + j1]);
}

// Tiny query graph (Nq=16, Eq=64): one block does everything in LDS.
__global__ __launch_bounds__(256)
void query_stage(const float* __restrict__ Xq, const int* __restrict__ eidx,
                 const float* __restrict__ W1a, const float* __restrict__ b1a,
                 const float* __restrict__ w2a, const float* __restrict__ b2a,
                 const float* __restrict__ W1v, const float* __restrict__ b1v,
                 const float* __restrict__ W2v, const float* __restrict__ b2v,
                 float* __restrict__ out, int Eq)
{
    __shared__ float sva[1024], svb[1024], shid[1024], svv[1024];
    int tid = threadIdx.x;
    for (int o = tid; o < 1024; o += 256) {
        int n = o >> 6, j = o & 63;
        float a0 = 0.0f, a1 = 0.0f, v0 = 0.0f;
        for (int k = 0; k < 64; ++k) {
            float x = Xq[n * 64 + k];
            a0 += x * W1a[k * 64 + j];
            a1 += x * W1a[(64 + k) * 64 + j];
            v0 += x * W1v[k * 64 + j];
        }
        sva[o] = a0;
        svb[o] = a1 + b1a[j];
        shid[o] = fmaxf(v0 + b1v[j], 0.0f);
    }
    __syncthreads();
    for (int o = tid; o < 1024; o += 256) {
        int n = o >> 6, j = o & 63;
        float v = b2v[j];
        for (int k = 0; k < 64; ++k) v += shid[n * 64 + k] * W2v[k * 64 + j];
        svv[o] = v;
    }
    __syncthreads();
    int wave = tid >> 5, lane = tid & 31;
    const int* src = eidx; const int* dst = eidx + Eq;
    for (int e = wave; e < Eq; e += 8) {
        int s = src[e], d = dst[e];
        int j0 = lane, j1 = lane + 32;
        float h0 = fmaxf(sva[s * 64 + j0] + svb[d * 64 + j0], 0.0f);
        float h1 = fmaxf(sva[s * 64 + j1] + svb[d * 64 + j1], 0.0f);
        float p = h0 * w2a[j0] + h1 * w2a[j1];
        for (int off = 16; off > 0; off >>= 1) p += __shfl_xor(p, off, 32);
        float att = 1.0f / (1.0f + __expf(-(p + b2a[0])));
        atomicAdd(&out[d * 64 + j0], att * svv[s * 64 + j0]);
        atomicAdd(&out[d * 64 + j1], att * svv[s * 64 + j1]);
    }
}

__global__ void zero_f32(float* __restrict__ p, size_t n) {
    size_t i = (size_t)blockIdx.x * blockDim.x + threadIdx.x;
    size_t st = (size_t)gridDim.x * blockDim.x;
    for (; i < n; i += st) p[i] = 0.0f;
}

extern "C" void kernel_launch(void* const* d_in, const int* in_sizes, int n_in,
                              void* d_out, int out_size, void* d_ws, size_t ws_size,
                              hipStream_t stream) {
    (void)n_in; (void)ws_size;
    const float* Xq   = (const float*)d_in[0];
    const int*   eq   = (const int*)d_in[1];
    const float* Xt   = (const float*)d_in[2];
    const int*   et   = (const int*)d_in[3];
    const unsigned char* mask = (const unsigned char*)d_in[4]; // jax bool = 1 byte
    const float* Wa1c = (const float*)d_in[5];   // [128,64]
    const float* ba1c = (const float*)d_in[6];
    const float* wa2c = (const float*)d_in[7];   // [64,1]  (logit bias cancels in softmax)
    const float* Wv1c = (const float*)d_in[9];   // [128,64]
    const float* bv1c = (const float*)d_in[10];
    const float* Wv2c = (const float*)d_in[11];  // [64,64]
    const float* bv2c = (const float*)d_in[12];
    const float* Wa1t = (const float*)d_in[13];  // [256,64]
    const float* ba1t = (const float*)d_in[14];
    const float* wa2t = (const float*)d_in[15];
    const float* ba2t = (const float*)d_in[16];
    const float* Wv1t = (const float*)d_in[17];  // [192,64]
    const float* bv1t = (const float*)d_in[18];
    const float* Wv2t = (const float*)d_in[19];
    const float* bv2t = (const float*)d_in[20];
    const float* Wa1q = (const float*)d_in[21];
    const float* ba1q = (const float*)d_in[22];
    const float* wa2q = (const float*)d_in[23];
    const float* ba2q = (const float*)d_in[24];
    const float* Wv1q = (const float*)d_in[25];
    const float* bv1q = (const float*)d_in[26];
    const float* Wv2q = (const float*)d_in[27];
    const float* bv2q = (const float*)d_in[28];

    const int D  = 64, Nq = 16;
    const int Nt = in_sizes[2] / D;     // 20000 (multiple of 16)
    const int Et = in_sizes[3] / 2;     // 320000
    const int Eq = in_sizes[1] / 2;     // 64

    float* out_q = (float*)d_out;            // Xq_new [16,64]
    float* out_t = (float*)d_out + Nq * D;   // Xt_new [Nt,64]

    // workspace (buffers reused across phases)
    char*  ws = (char*)d_ws;
    size_t NB = (size_t)Nt * 64 * sizeof(float);
    float*    bufA  = (float*)(ws);                 // HtA -> A1
    float*    bufB  = (float*)(ws + NB);            // HtV -> A2
    float*    bufC  = (float*)(ws + 2 * NB);        // VThid (f32 scratch)
    float*    bufD  = (float*)(ws + 3 * NB);        // Xqt -> VT
    _Float16* XtM   = (_Float16*)(ws + 4 * NB);     // [Nt,192] f16
    _Float16* xt_h  = (_Float16*)(ws + 4 * NB + (size_t)Nt * 192 * 2);
    _Float16* hbar_h= xt_h + (size_t)Nt * 64;       // Hbar f16 -> VThid f16 mirror
    _Float16* xq_h  = hbar_h + (size_t)Nt * 64;
    float*    HqA   = (float*)(xq_h + 1024);
    float*    HqV   = HqA + 1024;
    float*    Qv    = HqV + 1024;

    float* HtA = bufA;  float* HtV = bufB;
    float* A1  = bufA;  float* A2  = bufB;
    float* VThid = bufC; _Float16* VThid_h = hbar_h;
    float* Xqt = bufD;  float* VT = bufD;
    _Float16* no_h = xq_h;   // unused mirror target for MIRROR=false instantiations

    zero_f32<<<512, 256, 0, stream>>>((float*)d_out, (size_t)out_size);
    f32_to_f16<<<1, 256, 0, stream>>>(Xq, xq_h, (size_t)Nq * D);
    f32_to_f16<<<640, 256, 0, stream>>>(Xt, xt_h, (size_t)Nt * D);

    int gt = (Nt + 127) / 128;
    // cross-attention first layers, split across the concat
    gemm_n64<64, false, false><<<1,  256, 0, stream>>>(xq_h, Wa1c,           nullptr, HqA, no_h, Nq);
    gemm_n64<64, false, false><<<1,  256, 0, stream>>>(xq_h, Wv1c,           nullptr, HqV, no_h, Nq);
    gemm_n64<64, false, false><<<gt, 256, 0, stream>>>(xt_h, Wa1c + 64 * 64, ba1c,    HtA, no_h, Nt);
    gemm_n64<64, false, false><<<gt, 256, 0, stream>>>(xt_h, Wv1c + 64 * 64, bv1c,    HtV, no_h, Nt);
    cross_attention_kernel<<<(Nt + 7) / 8, 256, 0, stream>>>(HqA, HqV, HtA, HtV, wa2c, mask, hbar_h, Nt);
    gemm_n64<64, false, false><<<gt, 256, 0, stream>>>(hbar_h, Wv2c, bv2c, Xqt, no_h, Nt);
    colsum64<<<1, 256, 0, stream>>>(Xqt, Qv, Nt);
    build_xtm<<<640, 256, 0, stream>>>(Xt, Xqt, Qv, XtM, Nt);
    // target edge stage: per-node precompute then per-edge scatter
    gemm_n64<192, false, false><<<gt, 256, 0, stream>>>(XtM,     Wa1t,            ba1t,    A1,    no_h,    Nt);
    gemm_n64<64,  false, false><<<gt, 256, 0, stream>>>(xt_h,    Wa1t + 192 * 64, nullptr, A2,    no_h,    Nt);
    gemm_n64<192, true,  true ><<<gt, 256, 0, stream>>>(XtM,     Wv1t,            bv1t,    VThid, VThid_h, Nt);
    gemm_n64<64,  false, false><<<gt, 256, 0, stream>>>(VThid_h, Wv2t,            bv2t,    VT,    no_h,    Nt);
    edge_msg<<<(Et + 7) / 8, 256, 0, stream>>>(et, et + Et, A1, A2, VT, wa2t, ba2t, out_t, Et);
    // query graph (tiny)
    query_stage<<<1, 256, 0, stream>>>(Xq, eq, Wa1q, ba1q, wa2q, ba2q,
                                       Wv1q, bv1q, Wv2q, bv2q, out_q, Eq);
}
